// GATModel_60730837565929
// MI455X (gfx1250) — matrix operand order repaired and checked by reading, not compiled, not run on hardware
//
#include <hip/hip_runtime.h>
#include <hip/hip_bf16.h>

// ---------------------------------------------------------------------------
// GAT model for MI455X (gfx1250, wave32, WMMA).
// B=4, N=4096, D_IN=3072, H=64. Flash-style fused masked softmax per 16-row
// tile; all matmuls via v_wmma_f32_16x16x32_f16 (f16 operands, f32 accum).
// All f16 fragment traffic uses 128-bit loads; softmax stats are kept
// per-lane and merged across the 16-lane group once (deferred reduction).
// ---------------------------------------------------------------------------

#define NB 4
#define NN_ 4096
#define DIN 3072
#define HH 64
#define NTIL 256          // NN_/16 column/row tiles per batch

typedef _Float16 half16 __attribute__((ext_vector_type(16)));
typedef _Float16 h8v    __attribute__((ext_vector_type(8)));   // 16-byte chunk
typedef float    f32x8  __attribute__((ext_vector_type(8)));

__device__ __forceinline__ f32x8 wmma16(half16 a, half16 b, f32x8 c) {
  // D = A(16x32 f16) * B(32x16 f16) + C(16x16 f32)
  return __builtin_amdgcn_wmma_f32_16x16x32_f16(false, a, false, b,
                                                (short)0, c, false, false);
}

__device__ __forceinline__ f32x8 zero8() {
  f32x8 c;
#pragma unroll
  for (int i = 0; i < 8; ++i) c[i] = 0.0f;
  return c;
}

// 16 contiguous halves (32B) -> fragment (contiguous k order, used for A and B)
__device__ __forceinline__ half16 ld16h(const _Float16* p) {
  const h8v lo = *(const h8v*)p;
  const h8v hi = *(const h8v*)(p + 8);
  half16 r;
#pragma unroll
  for (int i = 0; i < 8; ++i) { r[i] = lo[i]; r[8 + i] = hi[i]; }
  return r;
}

// two separate 8-half chunks (16B each) -> fragment (A-layout chunks)
__device__ __forceinline__ half16 ld8h8h(const _Float16* p0, const _Float16* p1) {
  const h8v lo = *(const h8v*)p0;
  const h8v hi = *(const h8v*)p1;
  half16 r;
#pragma unroll
  for (int i = 0; i < 8; ++i) { r[i] = lo[i]; r[8 + i] = hi[i]; }
  return r;
}

// two 8-float chunks (f32, 32B each as 2x float4) -> f16 fragment
__device__ __forceinline__ half16 ld8f8f(const float* p0, const float* p1) {
  const float4 a0 = ((const float4*)p0)[0];
  const float4 a1 = ((const float4*)p0)[1];
  const float4 b0 = ((const float4*)p1)[0];
  const float4 b1 = ((const float4*)p1)[1];
  half16 r;
  r[0] = (_Float16)a0.x; r[1] = (_Float16)a0.y; r[2] = (_Float16)a0.z; r[3] = (_Float16)a0.w;
  r[4] = (_Float16)a1.x; r[5] = (_Float16)a1.y; r[6] = (_Float16)a1.z; r[7] = (_Float16)a1.w;
  r[8]  = (_Float16)b0.x; r[9]  = (_Float16)b0.y; r[10] = (_Float16)b0.z; r[11] = (_Float16)b0.w;
  r[12] = (_Float16)b1.x; r[13] = (_Float16)b1.y; r[14] = (_Float16)b1.z; r[15] = (_Float16)b1.w;
  return r;
}

// ---------------------------------------------------------------------------
// Encoder: Linear(3072->128) -> LayerNorm -> GELU(exact) -> Linear(128->64)
// grid=(4), block=(128). One block per batch element.
// ---------------------------------------------------------------------------
__global__ void enc_kernel(const float* __restrict__ stim,
                           const float* __restrict__ W1, const float* __restrict__ b1,
                           const float* __restrict__ lg, const float* __restrict__ lb,
                           const float* __restrict__ W2, const float* __restrict__ b2,
                           float* __restrict__ g_out) {
  const int b = blockIdx.x;
  const int j = threadIdx.x;           // 0..127
  __shared__ float sh[128];
  __shared__ float red[128];

  const float* srow = stim + (size_t)b * DIN;
  float acc = b1[j];
  for (int k = 0; k < DIN; ++k) acc = fmaf(srow[k], W1[(size_t)k * 128 + j], acc);
  sh[j] = acc; red[j] = acc;
  __syncthreads();
  for (int s = 64; s > 0; s >>= 1) { if (j < s) red[j] += red[j + s]; __syncthreads(); }
  const float mu = red[0] * (1.0f / 128.0f);
  __syncthreads();
  const float d = sh[j] - mu;
  red[j] = d * d;
  __syncthreads();
  for (int s = 64; s > 0; s >>= 1) { if (j < s) red[j] += red[j + s]; __syncthreads(); }
  const float var = red[0] * (1.0f / 128.0f);
  __syncthreads();
  float xn = d * rsqrtf(var + 1e-5f) * lg[j] + lb[j];
  float ge = 0.5f * xn * (1.0f + erff(xn * 0.70710678118654752f));  // exact GELU
  sh[j] = ge;
  __syncthreads();
  if (j < HH) {
    float a2 = b2[j];
    for (int k = 0; k < 128; ++k) a2 = fmaf(sh[k], W2[(size_t)k * HH + j], a2);
    g_out[b * HH + j] = a2;
  }
}

// x0[b,n,h] = g[b,h] + voxel_emb[n,h]
__global__ void x0_kernel(const float* __restrict__ g, const float* __restrict__ vemb,
                          float* __restrict__ x0) {
  const int idx = blockIdx.x * blockDim.x + threadIdx.x;   // < 4*4096*64
  const int h = idx & 63;
  const int n = (idx >> 6) & 4095;
  const int b = idx >> 18;
  x0[idx] = g[b * HH + h] + vemb[(size_t)n * HH + h];
}

// cm[col*256 + rt] : bit r set iff adj[rt*16+r][col] > 0
__global__ void colmask_kernel(const float* __restrict__ adj,
                               unsigned short* __restrict__ cm) {
  const int idx = blockIdx.x * blockDim.x + threadIdx.x;   // < 4096*256
  const int rt  = idx & 255;
  const int col = idx >> 8;
  unsigned m = 0;
#pragma unroll
  for (int r = 0; r < 16; ++r)
    if (adj[(size_t)(rt * 16 + r) * NN_ + col] > 0.0f) m |= (1u << r);
  cm[idx] = (unsigned short)m;
}

// ---------------------------------------------------------------------------
// Generic [16384,64] @ [64,64] GEMM via WMMA. Optional bias / addend / f32 out
// / f16 row-major out / f16 transposed-per-batch out.
// grid=(1024), block=(32) — one wave per 16-row tile.
// ---------------------------------------------------------------------------
__global__ void gemm64_kernel(const float* __restrict__ A, const float* __restrict__ W,
                              const float* __restrict__ bias, const float* __restrict__ addend,
                              float* __restrict__ outf, _Float16* __restrict__ oh,
                              _Float16* __restrict__ ohT) {
  const int lane = threadIdx.x;
  const int M  = lane & 15;
  const int hi = lane >> 4;
  const int rowbase = blockIdx.x * 16;

  // A fragments (16x32 f16 each; K map: i + 16*hi)
  const float* arow = A + (size_t)(rowbase + M) * HH;
  const half16 a0 = ld8f8f(arow +  0 + 16 * hi, arow +  8 + 16 * hi);
  const half16 a1 = ld8f8f(arow + 32 + 16 * hi, arow + 40 + 16 * hi);

  f32x8 acc[4];
#pragma unroll
  for (int ct = 0; ct < 4; ++ct) {
    half16 b0, b1;   // B fragment (32x16 f16): lane n = M, K = i + 16*hi (strided W)
#pragma unroll
    for (int i = 0; i < 16; ++i) {
      b0[i] = (_Float16)W[(size_t)( 0 + 16 * hi + i) * HH + ct * 16 + M];
      b1[i] = (_Float16)W[(size_t)(32 + 16 * hi + i) * HH + ct * 16 + M];
    }
    f32x8 c = zero8();
    c = wmma16(a0, b0, c);
    c = wmma16(a1, b1, c);
    acc[ct] = c;
  }

#pragma unroll
  for (int ct = 0; ct < 4; ++ct) {
#pragma unroll
    for (int r = 0; r < 8; ++r) {
      const int row = rowbase + r + 8 * hi;     // C layout: m=r+8*hi, n=lane&15
      const int col = ct * 16 + M;
      float v = acc[ct][r];
      if (bias)   v += bias[col];
      if (addend) v += addend[(size_t)row * HH + col];
      if (outf) outf[(size_t)row * HH + col] = v;
      if (oh)   oh[(size_t)row * HH + col] = (_Float16)v;
      if (ohT) {
        const int bb = row >> 12, n = row & 4095;
        ohT[((size_t)bb * HH + col) * NN_ + n] = (_Float16)v;
      }
    }
  }
}

// ---------------------------------------------------------------------------
// Fused GAT attention: scores = Wh·Whᵀ/8, mask, softmax, out = attn·Wh, (ELU).
// Two-pass flash softmax per 16-row tile; one wave per tile. Pass-1 softmax
// stats are per-lane (lane owns columns n ≡ lane (mod 16)); single deferred
// cross-lane (m,l) merge after the loop.
// grid=(1024), block=(32).
// ---------------------------------------------------------------------------
__global__ void gat_kernel(const _Float16* __restrict__ wh, const _Float16* __restrict__ whT,
                           const unsigned short* __restrict__ cm, float* __restrict__ out,
                           int do_elu) {
  const int lane = threadIdx.x;
  const int M  = lane & 15;
  const int hi = lane >> 4;
  const int b  = blockIdx.x >> 8;       // batch
  const int it = blockIdx.x & 255;      // row tile
  const size_t brow0 = (size_t)b * NN_;

  __shared__ __align__(16) _Float16 Pl[16 * 32];   // P tile: C-layout -> A-layout

  // Loop-invariant A fragments: this tile's 16 rows of Wh.
  const _Float16* arow = wh + (brow0 + it * 16 + M) * HH;
  const half16 a0 = ld16h(arow +      16 * hi);
  const half16 a1 = ld16h(arow + 32 + 16 * hi);

  float mrow[8], lrow[8];
#pragma unroll
  for (int r = 0; r < 8; ++r) { mrow[r] = -1e30f; lrow[r] = 0.0f; }

  // ---- Pass 1: per-lane online softmax statistics -----------------------
  for (int j = 0; j < NTIL; ++j) {
    const _Float16* brow = wh + (brow0 + j * 16 + M) * HH;
    if (j + 1 < NTIL)
      __builtin_prefetch(wh + (brow0 + (j + 1) * 16 + M) * HH, 0, 0);
    const half16 b0 = ld16h(brow +      16 * hi);
    const half16 b1 = ld16h(brow + 32 + 16 * hi);
    f32x8 c = zero8();
    c = wmma16(a0, b0, c);
    c = wmma16(a1, b1, c);
    const unsigned mk = cm[(size_t)(j * 16 + M) * NTIL + it];
#pragma unroll
    for (int r = 0; r < 8; ++r) {
      const bool on = (mk >> (r + 8 * hi)) & 1;
      const float se = on ? c[r] * 0.125f : -1e30f;     // / sqrt(H)
      const float mnew = fmaxf(mrow[r], se);
      lrow[r] = lrow[r] * __expf(mrow[r] - mnew) + (on ? __expf(se - mnew) : 0.0f);
      mrow[r] = mnew;
    }
  }
  // Deferred cross-lane merge of (m, l) within each 16-lane group.
  float invl[8];
#pragma unroll
  for (int r = 0; r < 8; ++r) {
    float m = mrow[r], l = lrow[r];
#pragma unroll
    for (int d2 = 1; d2 < 16; d2 <<= 1) {
      const float mo = __shfl_xor(m, d2, 16);
      const float lo = __shfl_xor(l, d2, 16);
      const float mn = fmaxf(m, mo);
      l = l * __expf(m - mn) + lo * __expf(mo - mn);
      m = mn;
    }
    mrow[r] = m;
    invl[r] = 1.0f / l;
  }

  f32x8 acc[4];
#pragma unroll
  for (int ct = 0; ct < 4; ++ct) acc[ct] = zero8();

  // ---- Pass 2: recompute scores -> P (f16) -> P @ Wh --------------------
  for (int jj = 0; jj < NTIL / 2; ++jj) {
#pragma unroll
    for (int t = 0; t < 2; ++t) {
      const int j = 2 * jj + t;
      const _Float16* brow = wh + (brow0 + j * 16 + M) * HH;
      if (j + 1 < NTIL)
        __builtin_prefetch(wh + (brow0 + (j + 1) * 16 + M) * HH, 0, 0);
      const half16 b0 = ld16h(brow +      16 * hi);
      const half16 b1 = ld16h(brow + 32 + 16 * hi);
      f32x8 c = zero8();
      c = wmma16(a0, b0, c);
      c = wmma16(a1, b1, c);
      const unsigned mk = cm[(size_t)(j * 16 + M) * NTIL + it];
#pragma unroll
      for (int r = 0; r < 8; ++r) {
        const bool on = (mk >> (r + 8 * hi)) & 1;
        const float s = c[r] * 0.125f;
        const float p = on ? __expf(s - mrow[r]) * invl[r] : 0.0f;
        Pl[(r + 8 * hi) * 32 + t * 16 + M] = (_Float16)p;   // [row m][K local]
      }
    }
    __syncthreads();
    // A-layout fragment of the 16x32 P tile (two 16B ds_load_b128)
    const half16 pa = ld16h(Pl + M * 32 + 16 * hi);
#pragma unroll
    for (int ct = 0; ct < 4; ++ct) {
      // B[k][n] = Wh[jj*32+k][ct*16+n] == whT[b][ct*16+n][jj*32+k] (contiguous)
      const _Float16* tcol = whT + ((size_t)(b * HH + ct * 16 + M)) * NN_ + jj * 32 + 16 * hi;
      if (jj + 1 < NTIL / 2)
        __builtin_prefetch(tcol + 32, 0, 0);
      acc[ct] = wmma16(pa, ld16h(tcol), acc[ct]);
    }
    __syncthreads();
  }

  // ---- Epilogue: optional ELU, store h' ---------------------------------
#pragma unroll
  for (int ct = 0; ct < 4; ++ct) {
#pragma unroll
    for (int r = 0; r < 8; ++r) {
      const int row = it * 16 + r + 8 * hi;
      float v = acc[ct][r];
      if (do_elu) v = v > 0.0f ? v : (__expf(v) - 1.0f);
      out[(brow0 + row) * HH + ct * 16 + M] = v;
    }
  }
}

// pred[b,n] = x2[b,n,:] . ro_W + ro_b
__global__ void readout_kernel(const float* __restrict__ x2, const float* __restrict__ roW,
                               const float* __restrict__ rob, float* __restrict__ pred) {
  const int idx = blockIdx.x * blockDim.x + threadIdx.x;   // < 16384
  const float* row = x2 + (size_t)idx * HH;
  float acc = rob[0];
#pragma unroll
  for (int h = 0; h < HH; ++h) acc = fmaf(row[h], roW[h], acc);
  pred[idx] = acc;
}

// ---------------------------------------------------------------------------
extern "C" void kernel_launch(void* const* d_in, const int* in_sizes, int n_in,
                              void* d_out, int out_size, void* d_ws, size_t ws_size,
                              hipStream_t stream) {
  (void)in_sizes; (void)n_in; (void)out_size; (void)ws_size;
  const float* stim = (const float*)d_in[0];
  const float* adj  = (const float*)d_in[1];
  const float* W1   = (const float*)d_in[2];
  const float* b1   = (const float*)d_in[3];
  const float* lg   = (const float*)d_in[4];
  const float* lb   = (const float*)d_in[5];
  const float* W2   = (const float*)d_in[6];
  const float* b2   = (const float*)d_in[7];
  const float* vemb = (const float*)d_in[8];
  const float* Wg1  = (const float*)d_in[9];
  const float* Wg2  = (const float*)d_in[10];
  const float* s1W  = (const float*)d_in[11];
  const float* s1b  = (const float*)d_in[12];
  const float* s2W  = (const float*)d_in[13];
  const float* s2b  = (const float*)d_in[14];
  const float* roW  = (const float*)d_in[15];
  const float* rob  = (const float*)d_in[16];
  float* pred = (float*)d_out;

  // Workspace layout (~23 MB total)
  char* ws = (char*)d_ws;
  const size_t SZX = (size_t)NB * NN_ * HH * sizeof(float);     // 4 MiB
  float*          g     = (float*)(ws);
  float*          x0    = (float*)(ws + 4096);
  float*          x1    = (float*)(ws + 4096 + 1 * SZX);
  float*          x2    = (float*)(ws + 4096 + 2 * SZX);
  float*          go    = (float*)(ws + 4096 + 3 * SZX);        // GAT output
  _Float16*       wh16  = (_Float16*)(ws + 4096 + 4 * SZX);     // 2 MiB
  _Float16*       wht16 = (_Float16*)(ws + 4096 + 4 * SZX + SZX / 2);
  unsigned short* cmask = (unsigned short*)(ws + 4096 + 4 * SZX + SZX);

  enc_kernel<<<NB, 128, 0, stream>>>(stim, W1, b1, lg, lb, W2, b2, g);
  x0_kernel<<<(NB * NN_ * HH) / 256, 256, 0, stream>>>(g, vemb, x0);
  colmask_kernel<<<(NN_ * NTIL) / 256, 256, 0, stream>>>(adj, cmask);

  // Layer 1
  gemm64_kernel<<<1024, 32, 0, stream>>>(x0, Wg1, nullptr, nullptr,
                                         nullptr, wh16, wht16);          // Wh1
  gat_kernel<<<1024, 32, 0, stream>>>(wh16, wht16, cmask, go, /*elu=*/1);
  gemm64_kernel<<<1024, 32, 0, stream>>>(x0, s1W, s1b, go,
                                         x1, nullptr, nullptr);          // x1 = gat + skip

  // Layer 2
  gemm64_kernel<<<1024, 32, 0, stream>>>(x1, Wg2, nullptr, nullptr,
                                         nullptr, wh16, wht16);          // Wh2
  gat_kernel<<<1024, 32, 0, stream>>>(wh16, wht16, cmask, go, /*elu=*/0);
  gemm64_kernel<<<1024, 32, 0, stream>>>(x1, s2W, s2b, go,
                                         x2, nullptr, nullptr);          // x2 = gat + skip

  readout_kernel<<<(NB * NN_) / 256, 256, 0, stream>>>(x2, roW, rob, pred);
}
